// ASpTLinear_33483565039896
// MI455X (gfx1250) — matrix operand order repaired and checked by reading, not compiled, not run on hardware
//
#include <hip/hip_runtime.h>

// ---------------------------------------------------------------------------
// out[m][n] = sum_k x[m][k] * W[n][k] + bias[n]
//   M = 8192 (B*S), N = 4096, K = 4096, fp32 in/out, bf16 WMMA compute.
//
// Fast path (needs 96 MiB workspace): pre-convert x and W to bf16 once, then
// run a double-buffered GEMM whose global->LDS stage uses CDNA5 async copies
// (global_load_async_to_lds_b128 + s_wait_asynccnt).
// Fallback path: fused fp32->bf16 conversion inside the GEMM (round-1 kernel).
// ---------------------------------------------------------------------------

typedef __attribute__((ext_vector_type(16))) __bf16 v16bf;
typedef __attribute__((ext_vector_type(8)))  float  v8f;
typedef __attribute__((ext_vector_type(4)))  int    v4i;

#define K_GLOB 4096
#define N_GLOB 4096
#define TM_BLK 128
#define TN_BLK 128
#define BK      64
#define LDT     72   // padded LDS row stride in bf16 elements (144 bytes)

union Frag {
    uint4 q[2];   // 32 bytes = 16 bf16
    v16bf v;
};

// fp32 -> bf16 (round-to-nearest-even), two at a time packed into a dword
__device__ __forceinline__ unsigned pack2_bf16(float a, float b) {
    unsigned ua = __float_as_uint(a);
    unsigned ub = __float_as_uint(b);
    ua = (ua + 0x7FFFu + ((ua >> 16) & 1u)) >> 16;
    ub = (ub + 0x7FFFu + ((ub >> 16) & 1u)) & 0xFFFF0000u;
    return ua | ub;
}

// ---------------------------------------------------------------------------
// CDNA5 async global->LDS helpers (ASYNCcnt-tracked)
// ---------------------------------------------------------------------------
typedef __attribute__((address_space(1))) v4i v4i_glob;
typedef __attribute__((address_space(3))) v4i v4i_lds;

__device__ __forceinline__ void async_b128(const unsigned short* g, unsigned short* l) {
#if __has_builtin(__builtin_amdgcn_global_load_async_to_lds_b128)
    __builtin_amdgcn_global_load_async_to_lds_b128(
        (v4i_glob*)g, (v4i_lds*)l, /*offset=*/0, /*cpol=*/0);
#else
    asm volatile("global_load_async_to_lds_b128 %0, %1, off"
                 :: "v"((unsigned)(unsigned long long)l), "v"(g) : "memory");
#endif
}

__device__ __forceinline__ void wait_async(int pending) {
#if __has_builtin(__builtin_amdgcn_s_wait_asynccnt)
    if (pending) __builtin_amdgcn_s_wait_asynccnt(8);
    else         __builtin_amdgcn_s_wait_asynccnt(0);
#else
    if (pending) asm volatile("s_wait_asynccnt 0x8" ::: "memory");
    else         asm volatile("s_wait_asynccnt 0x0" ::: "memory");
#endif
}

// ---------------------------------------------------------------------------
// fp32 -> bf16 bulk conversion (8 elements / thread)
// ---------------------------------------------------------------------------
__global__ __launch_bounds__(256)
void cvt_f32_to_bf16(const float* __restrict__ src,
                     unsigned short* __restrict__ dst, int n8) {
    int i = blockIdx.x * blockDim.x + threadIdx.x;
    if (i >= n8) return;
    const float4* s = (const float4*)src + 2 * (long)i;
    float4 a = s[0], b = s[1];
    uint4 q;
    q.x = pack2_bf16(a.x, a.y);
    q.y = pack2_bf16(a.z, a.w);
    q.z = pack2_bf16(b.x, b.y);
    q.w = pack2_bf16(b.z, b.w);
    ((uint4*)dst)[i] = q;
}

// ---------------------------------------------------------------------------
// Fast GEMM: bf16 inputs in global, async double-buffered LDS staging
// ---------------------------------------------------------------------------
__global__ __launch_bounds__(256)
void gemm_bf16_async(const unsigned short* __restrict__ Xb,
                     const unsigned short* __restrict__ Wb,
                     const float* __restrict__ bias,
                     float* __restrict__ out)
{
    __shared__ unsigned short sA[2][TM_BLK * LDT];
    __shared__ unsigned short sB[2][TN_BLK * LDT];

    const int tid  = threadIdx.x;
    const int lane = tid & 31;
    const int half = lane >> 4;
    const int l16  = lane & 15;
    const int wave = tid >> 5;
    const int waveM = wave >> 1;  // 0..3
    const int waveN = wave & 1;   // 0..1

    const int blockN = blockIdx.x * TN_BLK;
    const int blockM = blockIdx.y * TM_BLK;

    // loader mapping: 1 row x 32 bf16 (64 B = 4 x b128) per tile per thread
    const int lrow = tid >> 1;
    const int lcol = (tid & 1) * 32;

    const unsigned short* ga = Xb + (long)(blockM + lrow) * K_GLOB + lcol;
    const unsigned short* gb = Wb + (long)(blockN + lrow) * K_GLOB + lcol;
    const int ldsOff = lrow * LDT + lcol;

    v8f acc[2][4];
#pragma unroll
    for (int i = 0; i < 2; ++i)
#pragma unroll
        for (int j = 0; j < 4; ++j)
            acc[i][j] = (v8f)(0.0f);

    // prologue: async-copy tile 0 into buffer 0 (8 x b128 per thread)
#pragma unroll
    for (int c = 0; c < 4; ++c) async_b128(ga + 8 * c, &sA[0][ldsOff] + 8 * c);
#pragma unroll
    for (int c = 0; c < 4; ++c) async_b128(gb + 8 * c, &sB[0][ldsOff] + 8 * c);

    int buf = 0;
    for (int kb = 0; kb < K_GLOB; kb += BK) {
        const bool hasNext = (kb + BK) < K_GLOB;
        if (hasNext) {
            const unsigned short* na = ga + kb + BK;
            const unsigned short* nb = gb + kb + BK;
#pragma unroll
            for (int c = 0; c < 4; ++c)
                async_b128(na + 8 * c, &sA[buf ^ 1][ldsOff] + 8 * c);
#pragma unroll
            for (int c = 0; c < 4; ++c)
                async_b128(nb + 8 * c, &sB[buf ^ 1][ldsOff] + 8 * c);
        }
        wait_async(hasNext ? 8 : 0);  // this iteration's tile is resident
        __syncthreads();

#pragma unroll
        for (int ks = 0; ks < BK; ks += 32) {
            Frag a[2], b[4];
#pragma unroll
            for (int tm = 0; tm < 2; ++tm) {
                const int row = waveM * 32 + tm * 16 + l16;
                const unsigned short* base = &sA[buf][row * LDT + ks + half * 8];
                a[tm].q[0] = *(const uint4*)(base);
                a[tm].q[1] = *(const uint4*)(base + 16);
            }
#pragma unroll
            for (int tn = 0; tn < 4; ++tn) {
                const int row = waveN * 64 + tn * 16 + l16;
                const unsigned short* base = &sB[buf][row * LDT + ks + half * 8];
                b[tn].q[0] = *(const uint4*)(base);
                b[tn].q[1] = *(const uint4*)(base + 16);
            }
#pragma unroll
            for (int tm = 0; tm < 2; ++tm)
#pragma unroll
                for (int tn = 0; tn < 4; ++tn)
                    acc[tm][tn] = __builtin_amdgcn_wmma_f32_16x16x32_bf16(
                        false, a[tm].v, false, b[tn].v,
                        (short)0, acc[tm][tn], false, false);
        }
        __syncthreads();
        buf ^= 1;
    }

#pragma unroll
    for (int tn = 0; tn < 4; ++tn) {
        const int ncol = blockN + waveN * 64 + tn * 16 + l16;
        const float bv = bias[ncol];
#pragma unroll
        for (int tm = 0; tm < 2; ++tm) {
            const int mbase = blockM + waveM * 32 + tm * 16 + half * 8;
#pragma unroll
            for (int r = 0; r < 8; ++r)
                out[(long)(mbase + r) * N_GLOB + ncol] = acc[tm][tn][r] + bv;
        }
    }
}

// ---------------------------------------------------------------------------
// Fallback GEMM: fused fp32->bf16 conversion (no workspace needed)
// ---------------------------------------------------------------------------
__device__ __forceinline__ void cvt_store4(unsigned short* dst, const float4* f) {
    uint4* d = (uint4*)dst;
#pragma unroll
    for (int i = 0; i < 4; ++i) {
        uint4 q;
        q.x = pack2_bf16(f[2 * i].x,     f[2 * i].y);
        q.y = pack2_bf16(f[2 * i].z,     f[2 * i].w);
        q.z = pack2_bf16(f[2 * i + 1].x, f[2 * i + 1].y);
        q.w = pack2_bf16(f[2 * i + 1].z, f[2 * i + 1].w);
        d[i] = q;
    }
}

__global__ __launch_bounds__(256)
void gemm_fused_cvt(const float* __restrict__ X,
                    const float* __restrict__ W,
                    const float* __restrict__ bias,
                    float* __restrict__ out)
{
    __shared__ unsigned short sA[TM_BLK * LDT];
    __shared__ unsigned short sB[TN_BLK * LDT];

    const int tid  = threadIdx.x;
    const int lane = tid & 31;
    const int half = lane >> 4;
    const int l16  = lane & 15;
    const int wave = tid >> 5;
    const int waveM = wave >> 1;
    const int waveN = wave & 1;

    const int blockN = blockIdx.x * TN_BLK;
    const int blockM = blockIdx.y * TM_BLK;

    const int lrow = tid >> 1;
    const int lcol = (tid & 1) * 32;

    const float* ga = X + (long)(blockM + lrow) * K_GLOB + lcol;
    const float* gb = W + (long)(blockN + lrow) * K_GLOB + lcol;
    unsigned short* stA = &sA[lrow * LDT + lcol];
    unsigned short* stB = &sB[lrow * LDT + lcol];

    v8f acc[2][4];
#pragma unroll
    for (int i = 0; i < 2; ++i)
#pragma unroll
        for (int j = 0; j < 4; ++j)
            acc[i][j] = (v8f)(0.0f);

    for (int kb = 0; kb < K_GLOB; kb += BK) {
        float4 fa[8], fb[8];
        const float4* pa = (const float4*)(ga + kb);
        const float4* pb = (const float4*)(gb + kb);
#pragma unroll
        for (int i = 0; i < 8; ++i) { fa[i] = pa[i]; fb[i] = pb[i]; }

        if (kb + BK < K_GLOB) {
            __builtin_prefetch(ga + kb + BK, 0, 3);
            __builtin_prefetch(gb + kb + BK, 0, 3);
        }

        cvt_store4(stA, fa);
        cvt_store4(stB, fb);
        __syncthreads();

#pragma unroll
        for (int ks = 0; ks < BK; ks += 32) {
            Frag a[2], b[4];
#pragma unroll
            for (int tm = 0; tm < 2; ++tm) {
                const int row = waveM * 32 + tm * 16 + l16;
                const unsigned short* base = &sA[row * LDT + ks + half * 8];
                a[tm].q[0] = *(const uint4*)(base);
                a[tm].q[1] = *(const uint4*)(base + 16);
            }
#pragma unroll
            for (int tn = 0; tn < 4; ++tn) {
                const int row = waveN * 64 + tn * 16 + l16;
                const unsigned short* base = &sB[row * LDT + ks + half * 8];
                b[tn].q[0] = *(const uint4*)(base);
                b[tn].q[1] = *(const uint4*)(base + 16);
            }
#pragma unroll
            for (int tm = 0; tm < 2; ++tm)
#pragma unroll
                for (int tn = 0; tn < 4; ++tn)
                    acc[tm][tn] = __builtin_amdgcn_wmma_f32_16x16x32_bf16(
                        false, a[tm].v, false, b[tn].v,
                        (short)0, acc[tm][tn], false, false);
        }
        __syncthreads();
    }

#pragma unroll
    for (int tn = 0; tn < 4; ++tn) {
        const int ncol = blockN + waveN * 64 + tn * 16 + l16;
        const float bv = bias[ncol];
#pragma unroll
        for (int tm = 0; tm < 2; ++tm) {
            const int mbase = blockM + waveM * 32 + tm * 16 + half * 8;
#pragma unroll
            for (int r = 0; r < 8; ++r)
                out[(long)(mbase + r) * N_GLOB + ncol] = acc[tm][tn][r] + bv;
        }
    }
}

// ---------------------------------------------------------------------------
extern "C" void kernel_launch(void* const* d_in, const int* in_sizes, int n_in,
                              void* d_out, int out_size, void* d_ws, size_t ws_size,
                              hipStream_t stream) {
    const float* X    = (const float*)d_in[0];  // [M, 4096] fp32
    const float* W    = (const float*)d_in[1];  // [4096, 4096] fp32 (masked)
    const float* bias = (const float*)d_in[2];  // [4096] fp32
    float* out        = (float*)d_out;          // [M, 4096] fp32

    const int M = in_sizes[0] / K_GLOB;         // 8192
    const size_t xElems = (size_t)M * K_GLOB;
    const size_t wElems = (size_t)N_GLOB * K_GLOB;
    const size_t needWs = (xElems + wElems) * sizeof(unsigned short);  // 96 MiB

    dim3 grid(N_GLOB / TN_BLK, M / TM_BLK);
    dim3 block(256);

    if (ws_size >= needWs) {
        unsigned short* Xb = (unsigned short*)d_ws;
        unsigned short* Wb = Xb + xElems;
        const int xN8 = (int)(xElems / 8);
        const int wN8 = (int)(wElems / 8);
        hipLaunchKernelGGL(cvt_f32_to_bf16, dim3((xN8 + 255) / 256), block, 0,
                           stream, X, Xb, xN8);
        hipLaunchKernelGGL(cvt_f32_to_bf16, dim3((wN8 + 255) / 256), block, 0,
                           stream, W, Wb, wN8);
        hipLaunchKernelGGL(gemm_bf16_async, grid, block, 0, stream,
                           Xb, Wb, bias, out);
    } else {
        hipLaunchKernelGGL(gemm_fused_cvt, grid, block, 0, stream,
                           X, W, bias, out);
    }
}
